// ProjectiveSAGEConv_69406671503586
// MI455X (gfx1250) — matrix operand order repaired and checked by reading, not compiled, not run on hardware
//
#include <hip/hip_runtime.h>
#include <math.h>

typedef __attribute__((ext_vector_type(16))) _Float16 v16h;
typedef __attribute__((ext_vector_type(8)))  float    v8f;

#define EPSF 1e-8f

// ---------------------------------------------------------------------------
// Zero a float buffer (graph-capture-safe replacement for hipMemsetAsync).
// ---------------------------------------------------------------------------
__global__ __launch_bounds__(256)
void zero_kernel(float* __restrict__ p, int n) {
    int i = blockIdx.x * 256 + threadIdx.x;
    if (i < n) p[i] = 0.0f;
}

// ---------------------------------------------------------------------------
// C[m][o] = sum_k X[m][k] * W[o][k]   (X: N x 64, W: 64 x 64 row-major)
// One block = 128 threads = 4 waves; block handles a 16-row tile, each wave
// produces one 16x16 output tile via two chained v_wmma_f32_16x16x32_f16.
// A/B staged through LDS as f16.
// ---------------------------------------------------------------------------
__global__ __launch_bounds__(128)
void gemm_n64_wmma(const float* __restrict__ X, const float* __restrict__ W,
                   float* __restrict__ out) {
    __shared__ _Float16 sA[16 * 64];   // x tile, row-major [m][k]
    __shared__ _Float16 sB[64 * 64];   // W as-is, [o][k]  (B[k][n] == W[n][k])

    const int tid   = threadIdx.x;
    const int tile0 = blockIdx.x * 16;

    // Stage x tile (1024 f32) and W (4096 f32) into LDS as f16, coalesced.
    #pragma unroll
    for (int idx = tid; idx < 16 * 64; idx += 128)
        sA[idx] = (_Float16)X[tile0 * 64 + idx];
    for (int idx = tid; idx < 64 * 64; idx += 128)
        sB[idx] = (_Float16)W[idx];
    __syncthreads();

    const int lane = tid & 31;
    const int wave = tid >> 5;
    const int h    = lane >> 4;    // lane half (0: lanes 0-15, 1: lanes 16-31)
    const int q    = lane & 15;    // A row index / B,C column index
    const int n0   = wave * 16;    // column tile base

    // A-matrix layout (16-bit A 16x32): m = lane&15, element e -> k = 16h+e.
    v16h a_lo, a_hi;
    #pragma unroll
    for (int e = 0; e < 16; ++e) {
        int j = e >> 1, p = e & 1;
        int k = 16 * h + e; (void)j; (void)p;
        a_lo[e] = sA[q * 64 + k];
        a_hi[e] = sA[q * 64 + 32 + k];
    }

    // B-matrix layout (16-bit B 32x16): n = lane&15, element e -> k = 16h+e.
    const _Float16* bp = &sB[(n0 + q) * 64 + 16 * h];
    v16h b_lo, b_hi;
    #pragma unroll
    for (int e = 0; e < 16; ++e) {
        b_lo[e] = bp[e];
        b_hi[e] = bp[32 + e];
    }

    v8f c = {};
    // (neg_a, A, neg_b, B, c_mod, C, reuse_a, reuse_b)
    c = __builtin_amdgcn_wmma_f32_16x16x32_f16(false, a_lo, false, b_lo,
                                               (short)0, c, false, false);
    c = __builtin_amdgcn_wmma_f32_16x16x32_f16(false, a_hi, false, b_hi,
                                               (short)0, c, false, false);

    // C/D layout: VGPR r -> row = r + 8h, col = lane&15.
    #pragma unroll
    for (int r = 0; r < 8; ++r)
        out[(tile0 + r + 8 * h) * 64 + n0 + q] = c[r];
}

// ---------------------------------------------------------------------------
// In-place row L2-normalize of an N x 64 matrix. One wave per row,
// 2 elements per lane, wave32 butterfly reduction.
// ---------------------------------------------------------------------------
__global__ __launch_bounds__(256)
void rownorm_inplace(float* __restrict__ v, int N) {
    int node = (blockIdx.x * 256 + threadIdx.x) >> 5;
    int lane = threadIdx.x & 31;
    if (node >= N) return;
    float* r = v + (size_t)node * 64;
    float a = r[2 * lane], b = r[2 * lane + 1];
    float ss = a * a + b * b;
    #pragma unroll
    for (int o = 16; o; o >>= 1) ss += __shfl_xor(ss, o, 32);
    float inv = 1.0f / (sqrtf(ss) + EPSF);
    r[2 * lane]     = a * inv;
    r[2 * lane + 1] = b * inv;
}

// ---------------------------------------------------------------------------
// Edge pass: one wave per edge. Cosine of raw features -> clipped weight w.
// agg[row] += w * ndir[col]; agg[col] += w * ndir[row]  (atomic f32 adds).
// ---------------------------------------------------------------------------
__global__ __launch_bounds__(256)
void edge_aggregate(const int* __restrict__ ei, const float* __restrict__ X,
                    const float* __restrict__ ndir, float* __restrict__ agg,
                    int E) {
    int e    = (blockIdx.x * 256 + threadIdx.x) >> 5;   // uniform in wave
    int lane = threadIdx.x & 31;
    if (e >= E) return;

    int r = ei[e];
    int c = ei[E + e];

    const float* ps = X + (size_t)r * 64;
    const float* pd = X + (size_t)c * 64;
    float s0 = ps[2 * lane], s1 = ps[2 * lane + 1];
    float d0 = pd[2 * lane], d1 = pd[2 * lane + 1];

    float dot = s0 * d0 + s1 * d1;
    float na  = s0 * s0 + s1 * s1;
    float nb  = d0 * d0 + d1 * d1;
    #pragma unroll
    for (int o = 16; o; o >>= 1) {
        dot += __shfl_xor(dot, o, 32);
        na  += __shfl_xor(na,  o, 32);
        nb  += __shfl_xor(nb,  o, 32);
    }
    float cosv = dot / (sqrtf(na) * sqrtf(nb) + EPSF);
    float w    = fminf(fmaxf(0.5f * (cosv + 1.0f), 0.1f), 0.9f);

    const float* nr = ndir + (size_t)r * 64;
    const float* nc = ndir + (size_t)c * 64;
    float* ar = agg + (size_t)r * 64;
    float* ac = agg + (size_t)c * 64;
    atomicAdd(&ar[2 * lane],     w * nc[2 * lane]);
    atomicAdd(&ar[2 * lane + 1], w * nc[2 * lane + 1]);
    atomicAdd(&ac[2 * lane],     w * nr[2 * lane]);
    atomicAdd(&ac[2 * lane + 1], w * nr[2 * lane + 1]);
}

// ---------------------------------------------------------------------------
// Finalize: one wave per node.
//   n = normalize(ndir[i] + agg[i])           (diag contribution + neighbors)
//   o = normalize(sdir[i] + n)
//   out = normalize(0.9*o + 0.1*bias/||[bias;1]||)
// sdir aliases d_out; each lane reads its own slots before writing them.
// ---------------------------------------------------------------------------
__global__ __launch_bounds__(256)
void finalize_kernel(const float* __restrict__ sdir, const float* __restrict__ ndir,
                     const float* __restrict__ agg, const float* __restrict__ bias,
                     float* __restrict__ out, int N) {
    int node = (blockIdx.x * 256 + threadIdx.x) >> 5;
    int lane = threadIdx.x & 31;
    if (node >= N) return;
    size_t base = (size_t)node * 64;
    int i0 = 2 * lane, i1 = 2 * lane + 1;

    // neighbor direction
    float v0 = ndir[base + i0] + agg[base + i0];
    float v1 = ndir[base + i1] + agg[base + i1];
    float ss = v0 * v0 + v1 * v1;
    #pragma unroll
    for (int o = 16; o; o >>= 1) ss += __shfl_xor(ss, o, 32);
    float inv = 1.0f / (sqrtf(ss) + EPSF);
    float n0 = v0 * inv, n1 = v1 * inv;

    // blend with self direction
    float o0 = sdir[base + i0] + n0;
    float o1 = sdir[base + i1] + n1;
    ss = o0 * o0 + o1 * o1;
    #pragma unroll
    for (int o = 16; o; o >>= 1) ss += __shfl_xor(ss, o, 32);
    inv = 1.0f / (sqrtf(ss) + EPSF);
    o0 *= inv; o1 *= inv;

    // bias point direction: bias / sqrt(||bias||^2 + 1)
    float b0 = bias[i0], b1 = bias[i1];
    float sb = b0 * b0 + b1 * b1;
    #pragma unroll
    for (int o = 16; o; o >>= 1) sb += __shfl_xor(sb, o, 32);
    float binv = 1.0f / sqrtf(sb + 1.0f);

    float f0 = 0.9f * o0 + 0.1f * b0 * binv;
    float f1 = 0.9f * o1 + 0.1f * b1 * binv;
    ss = f0 * f0 + f1 * f1;
    #pragma unroll
    for (int o = 16; o; o >>= 1) ss += __shfl_xor(ss, o, 32);
    inv = 1.0f / (sqrtf(ss) + EPSF);

    out[base + i0] = f0 * inv;
    out[base + i1] = f1 * inv;
}

// ---------------------------------------------------------------------------
extern "C" void kernel_launch(void* const* d_in, const int* in_sizes, int n_in,
                              void* d_out, int out_size, void* d_ws, size_t ws_size,
                              hipStream_t stream) {
    const float* x      = (const float*)d_in[0];   // N x 64
    const int*   ei     = (const int*)  d_in[1];   // 2 x E
    const float* Wself  = (const float*)d_in[2];   // 64 x 64
    const float* Wneigh = (const float*)d_in[3];   // 64 x 64
    const float* bias   = (const float*)d_in[4];   // 64

    const int N = in_sizes[0] / 64;
    const int E = in_sizes[1] / 2;

    float* out  = (float*)d_out;          // also holds self projection
    float* ndir = (float*)d_ws;           // N*64 f32: neighbor projection dirs
    float* agg  = ndir + (size_t)N * 64;  // N*64 f32: atomic accumulator

    const int nwave_blocks = (N * 32 + 255) / 256;

    zero_kernel<<<(N * 64 + 255) / 256, 256, 0, stream>>>(agg, N * 64);

    gemm_n64_wmma<<<N / 16, 128, 0, stream>>>(x, Wself,  out);
    gemm_n64_wmma<<<N / 16, 128, 0, stream>>>(x, Wneigh, ndir);

    rownorm_inplace<<<nwave_blocks, 256, 0, stream>>>(out,  N);
    rownorm_inplace<<<nwave_blocks, 256, 0, stream>>>(ndir, N);

    edge_aggregate<<<(E * 32 + 255) / 256, 256, 0, stream>>>(ei, x, ndir, agg, E);

    finalize_kernel<<<nwave_blocks, 256, 0, stream>>>(out, ndir, agg, bias, out, N);
}
